// CovidsV2_15633680957949
// MI455X (gfx1250) — matrix operand order repaired and checked by reading, not compiled, not run on hardware
//
#include <hip/hip_runtime.h>

#define Bsz 256
#define Lsz 336
#define Dsz 32
#define Tsz 12
#define Msz 28
#define PREDsz 8

typedef __attribute__((ext_vector_type(16))) __bf16 v16bf;
typedef __attribute__((ext_vector_type(8)))  float  v8f;
typedef union { v8f v; float f[8]; } v8f_u;

// ---------------- ws layout (float words) ----------------
#define WSF_SIGMA   0                        // 7168*1024
#define WSF_MEAN    (7168*1024)              // 7168*32
#define WSF_MEANALL (WSF_MEAN + 7168*32)     // 256*32
#define WSF_SX      (WSF_MEANALL + 256*32)   // 256*384  [v][k] per b
#define WSF_END     (WSF_SX + 256*384)
// bf16 weight region, offsets in *uint32* units from (uint*)(wsf+WSF_END)
#define WBU_WMIN   0         // 512 x 32  (K pad 28->32), stride 16 uints
#define WBU_WMRAW  8192      // 512 x 352 (336->352),     stride 176
#define WBU_WMOUT  98304     // 16  x 1024 (rows 8..15 zero), stride 512
#define WBU_WSIN   106496    // 512 x 32, stride 16
#define WBU_W12    114688    // 512 x 160 (144->160), stride 80
#define WBU_WSOUT  155648    // 16 x 1024, stride 512

#define OFF_PM 2752512       // x_norm = 256*336*32
#define OFF_PS 2818048       // + preds_mean 256*8*32

// ---------------- helpers ----------------
__device__ inline unsigned short f2bf(float x) {
  unsigned u = __float_as_uint(x);
  u = (u + 0x7FFFu + ((u >> 16) & 1u)) >> 16;
  return (unsigned short)u;
}
__device__ inline unsigned pack_bf16(float a, float b) {
  return (unsigned)f2bf(a) | ((unsigned)f2bf(b) << 16);
}
// A fragment: 16x32 bf16 tile, row-major, strideU uints per row.
// lane: m = lane&15, h = lane>>4; VGPR i<4: K=h*8+2i ; i>=4: K=16+h*8+2(i-4)
__device__ inline void load_fragA(const unsigned* base, int strideU, int lane, unsigned u[8]) {
  int r = lane & 15, h = lane >> 4;
  const unsigned* p = base + r * strideU + h * 4;
  u[0]=p[0]; u[1]=p[1]; u[2]=p[2]; u[3]=p[3];
  u[4]=p[8]; u[5]=p[9]; u[6]=p[10]; u[7]=p[11];
}
// A fragment built on-the-fly from f32 LDS (optionally fused relu)
__device__ inline void load_fragA_f32(const float* base, int strideF, int lane,
                                      unsigned u[8], bool relu) {
  int r = lane & 15, h = lane >> 4;
  const float* p = base + r * strideF + h * 8;
#pragma unroll
  for (int i = 0; i < 4; i++) {
    float a = p[2*i], b = p[2*i+1], c = p[16+2*i], d = p[16+2*i+1];
    if (relu) { a=fmaxf(a,0.f); b=fmaxf(b,0.f); c=fmaxf(c,0.f); d=fmaxf(d,0.f); }
    u[i]   = pack_bf16(a, b);
    u[4+i] = pack_bf16(c, d);
  }
}
// B fragment: B[k][n] = W[n][k]; W row-major [N][Kpad], strideU uints.
// lane: n = lane&15, h = lane>>4; VGPR i<4: K=h*8+2i ; i>=4: K=16+h*8+2(i-4)
__device__ inline void load_fragB(const unsigned* base, int strideU, int lane, unsigned u[8]) {
  int n = lane & 15, h = lane >> 4;
  const unsigned* p = base + n * strideU + h * 4;
#pragma unroll
  for (int i = 0; i < 4; i++) { u[i] = p[i]; u[4+i] = p[8+i]; }
}
__device__ inline v8f wmma_bf16(const unsigned a[8], const unsigned b[8], v8f c) {
  union { v16bf v; unsigned u[8]; } A, Bm;
#pragma unroll
  for (int i = 0; i < 8; i++) { A.u[i] = a[i]; Bm.u[i] = b[i]; }
  return __builtin_amdgcn_wmma_f32_16x16x32_bf16(false, A.v, false, Bm.v,
                                                 (short)0, c, false, false);
}

// ---------------- weight prep (f32 -> bf16 padded layouts) ----------------
__global__ void k_prep_weights(const float* wm_in, const float* wm_raw, const float* wm_out,
                               const float* ws_in, const float* ws_r1, const float* ws_r2,
                               const float* ws_out, unsigned short* wb) {
  int tid = blockIdx.x * blockDim.x + threadIdx.x;
  int str = gridDim.x * blockDim.x;
  for (int i = tid; i < 512*32; i += str) { int t=i>>5,k=i&31;
    wb[2*WBU_WMIN + i]  = f2bf(k < 28 ? wm_in[t*28+k] : 0.f); }
  for (int i = tid; i < 512*352; i += str) { int t=i/352,k=i%352;
    wb[2*WBU_WMRAW + i] = f2bf(k < 336 ? wm_raw[t*336+k] : 0.f); }
  for (int i = tid; i < 16*1024; i += str) { int n=i>>10,k=i&1023;
    wb[2*WBU_WMOUT + i] = f2bf(n < 8 ? wm_out[n*1024+k] : 0.f); }
  for (int i = tid; i < 512*32; i += str) { int t=i>>5,k=i&31;
    wb[2*WBU_WSIN + i]  = f2bf(k < 28 ? ws_in[t*28+k] : 0.f); }
  for (int i = tid; i < 512*160; i += str) { int t=i/160,k=i%160;
    float v = 0.f;
    if (k < 144) { int k1=k/12, k2=k%12; v = ws_r1[t*12+k1]*ws_r2[t*12+k2]; }
    wb[2*WBU_W12 + i] = f2bf(v); }
  for (int i = tid; i < 16*1024; i += str) { int n=i>>10,k=i&1023;
    wb[2*WBU_WSOUT + i] = f2bf(n < 8 ? ws_out[n*1024+k] : 0.f); }
}

// ---------------- per-(b,m) mean + covariance ----------------
__global__ __launch_bounds__(128) void k_stats(const float* bx, float* wsf) {
  int bm = blockIdx.x; int b = bm / 28, m = bm % 28;
  __shared__ float xc[12][32];
  __shared__ float mu[32];
  int tid = threadIdx.x;
  for (int i = tid; i < 384; i += 128) {
    int t = i >> 5, v = i & 31;
    xc[t][v] = bx[(size_t)b*10752 + (m*12+t)*32 + v];
  }
  __syncthreads();
  if (tid < 32) {
    float s = 0.f;
    for (int t = 0; t < 12; t++) s += xc[t][tid];
    mu[tid] = s * (1.f/12.f);
    wsf[WSF_MEAN + bm*32 + tid] = mu[tid];
  }
  __syncthreads();
  for (int i = tid; i < 384; i += 128) { int t=i>>5, v=i&31; xc[t][v] -= mu[v]; }
  __syncthreads();
  for (int i = tid; i < 1024; i += 128) {
    int r = i >> 5, c = i & 31;
    float s = 0.f;
    for (int t = 0; t < 12; t++) s += xc[t][r] * xc[t][c];
    wsf[WSF_SIGMA + (size_t)bm*1024 + i] = s * (1.f/11.f);
  }
}

// ---------------- per-b mean_all and Sx[v][k] = sum_m x[b, m*12+k, v] ----------------
__global__ __launch_bounds__(128) void k_meanall_sx(const float* bx, float* wsf) {
  int b = blockIdx.x;
  __shared__ float sx[32][12];
  int tid = threadIdx.x;
  for (int i = tid; i < 384; i += 128) {
    int v = i / 12, k = i % 12;
    float s = 0.f;
    for (int m = 0; m < 28; m++) s += bx[(size_t)b*10752 + (m*12+k)*32 + v];
    sx[v][k] = s;
    wsf[WSF_SX + b*384 + i] = s;
  }
  __syncthreads();
  if (tid < 32) {
    float s = 0.f;
    for (int k = 0; k < 12; k++) s += sx[tid][k];
    wsf[WSF_MEANALL + b*32 + tid] = s * (1.f/336.f);
  }
}

// ---------------- Jacobi eigensolver (one wave32 per 32x32 Sigma) + whitening ----------------
__global__ __launch_bounds__(32) void k_eigen(const float* bx, float* wsf, float* dout) {
  int bm = blockIdx.x; int b = bm / 28, m = bm % 28;
  __shared__ float A[32][33], V[32][33];
  __shared__ float cs[16], sn[16];
  __shared__ int   pp[16], qq[16];
  __shared__ float dg[32], isvv[32];
  __shared__ int   inv[32];
  __shared__ float xcs[12][32];
  int l = threadIdx.x;
  for (int j = 0; j < 32; j++) A[l][j] = wsf[WSF_SIGMA + (size_t)bm*1024 + l*32 + j];
  for (int j = 0; j < 32; j++) V[l][j] = (l == j) ? 1.f : 0.f;
  __syncthreads();
  for (int sweep = 0; sweep < 8; sweep++) {
    for (int r = 0; r < 31; r++) {
      if (l < 16) {
        int p, q;
        if (l == 0) { p = 31; q = r; }
        else        { p = (r + l) % 31; q = (r - l + 31) % 31; }
        if (p > q) { int t = p; p = q; q = t; }
        float apq = A[p][q], app = A[p][p], aqq = A[q][q];
        float c = 1.f, s = 0.f;
        if (fabsf(apq) > 1e-20f) {
          float tau = (aqq - app) / (2.f * apq);
          float t = (tau >= 0.f ? 1.f : -1.f) / (fabsf(tau) + sqrtf(1.f + tau*tau));
          c = rsqrtf(1.f + t*t); s = t * c;
        }
        cs[l] = c; sn[l] = s; pp[l] = p; qq[l] = q;
      }
      __syncthreads();
      for (int i = 0; i < 16; i++) {      // row update; lane l owns column l
        int p = pp[i], q = qq[i]; float c = cs[i], s = sn[i];
        float a = A[p][l], bb = A[q][l];
        A[p][l] = c*a - s*bb; A[q][l] = s*a + c*bb;
      }
      __syncthreads();
      for (int i = 0; i < 16; i++) {      // column update; lane l owns row l
        int p = pp[i], q = qq[i]; float c = cs[i], s = sn[i];
        float a = A[l][p], bb = A[l][q];
        A[l][p] = c*a - s*bb; A[l][q] = s*a + c*bb;
        float va = V[l][p], vb = V[l][q];
        V[l][p] = c*va - s*vb; V[l][q] = s*va + c*vb;
      }
      __syncthreads();
    }
  }
  dg[l] = A[l][l];
  __syncthreads();
  int rank = 0;
  for (int u = 0; u < 32; u++) {
    float du = dg[u];
    if (du > dg[l] || (du == dg[l] && u < l)) rank++;
  }
  inv[rank] = l;
  __syncthreads();
  isvv[l] = rsqrtf(fmaxf(dg[inv[l]], 0.f) + 1e-5f);
  for (int j = 0; j < 32; j++) A[l][j] = V[l][inv[j]];   // U sorted desc (row l)
  float muv = wsf[WSF_MEAN + bm*32 + l];
  for (int t = 0; t < 12; t++)
    xcs[t][l] = bx[(size_t)b*10752 + (m*12+t)*32 + l] - muv;
  __syncthreads();
  for (int t = 0; t < 12; t++) {          // x_norm[t][v=l] = isv[v] * sum_n U[v][n] xc[t][n]
    float s = 0.f;
    for (int n = 0; n < 32; n++) s += A[l][n] * xcs[t][n];
    dout[(size_t)b*10752 + (m*12+t)*32 + l] = s * isvv[l];
  }
}

// ---------------- fused mean-branch MLP (WMMA) ----------------
__global__ __launch_bounds__(128) void k_mean_branch(const float* bx, const float* bm_in,
    const float* bm_raw, const float* bm_out, const float* weight,
    const float* wsf, const unsigned* wbU, float* dout) {
  extern __shared__ char smem[];
  unsigned* A1u = (unsigned*)smem;        // 16x16  uints (16x32 bf16)
  unsigned* A2u = A1u + 256;              // 16x176 uints (16x352 bf16)
  float*    h   = (float*)(A2u + 2816);   // 16x1024 f32
  float*    PD  = h + 16384;              // 4x16x16
  int rt = blockIdx.x; int b = rt >> 1; int vbase = (rt & 1) * 16;
  int tid = threadIdx.x, lane = tid & 31, wid = tid >> 5;
  const float* meanall = wsf + WSF_MEANALL + b*32;
  for (int i = tid; i < 256; i += 128) {
    int mr = i >> 4, kp = i & 15; int v = vbase + mr; float ma = meanall[v];
    int k0 = kp*2, k1 = k0 + 1;
    float a = (k0 < 28) ? wsf[WSF_MEAN + (b*28+k0)*32 + v] - ma : 0.f;
    float c = (k1 < 28) ? wsf[WSF_MEAN + (b*28+k1)*32 + v] - ma : 0.f;
    A1u[mr*16 + kp] = pack_bf16(a, c);
  }
  for (int i = tid; i < 2816; i += 128) {
    int mr = i / 176, kp = i % 176; int v = vbase + mr; float ma = meanall[v];
    int k0 = kp*2, k1 = k0 + 1;
    float a = (k0 < 336) ? bx[(size_t)b*10752 + k0*32 + v] - ma : 0.f;
    float c = (k1 < 336) ? bx[(size_t)b*10752 + k1*32 + v] - ma : 0.f;
    A2u[mr*176 + kp] = pack_bf16(a, c);
  }
  __syncthreads();
  int n = lane & 15, hh = lane >> 4;
  // hoist all A fragments (nt-invariant) into registers
  unsigned af1[8];
  load_fragA(A1u, 16, lane, af1);
  unsigned afr[11][8];
#pragma unroll
  for (int kt = 0; kt < 11; kt++) load_fragA(A2u + kt*16, 176, lane, afr[kt]);
  for (int nt = wid; nt < 32; nt += 4) {
    // preload all B fragments for this nt, then issue WMMAs back-to-back
    unsigned bf1[8];
    load_fragB(wbU + WBU_WMIN + nt*16*16, 16, lane, bf1);
    unsigned bArr[11][8];
#pragma unroll
    for (int kt = 0; kt < 11; kt++)
      load_fragB(wbU + WBU_WMRAW + nt*16*176 + kt*16, 176, lane, bArr[kt]);
    v8f_u acc; acc.v = (v8f){};
    acc.v = wmma_bf16(af1, bf1, acc.v);
#pragma unroll
    for (int r = 0; r < 8; r++) h[(r + 8*hh)*1024 + nt*16 + n] = acc.f[r] + bm_in[nt*16 + n];
    v8f_u acc2; acc2.v = (v8f){};
#pragma unroll
    for (int kt = 0; kt < 11; kt++)
      acc2.v = wmma_bf16(afr[kt], bArr[kt], acc2.v);
#pragma unroll
    for (int r = 0; r < 8; r++) h[(r + 8*hh)*1024 + 512 + nt*16 + n] = acc2.f[r] + bm_raw[nt*16 + n];
  }
  __syncthreads();
  // output GEMM: relu(h) @ wm_out^T, split over 4 waves; B preloaded
  unsigned bouts[8][8];
  {
    int c = 0;
#pragma unroll
    for (int c2 = 0; c2 < 8; c2++) {
      int kt = wid + 4*c2;
      load_fragB(wbU + WBU_WMOUT + kt*16, 512, lane, bouts[c2]);
      c = c2;
    }
    (void)c;
  }
  v8f_u acc; acc.v = (v8f){};
#pragma unroll
  for (int c2 = 0; c2 < 8; c2++) {
    int kt = wid + 4*c2;
    unsigned af[8];
    load_fragA_f32(h + kt*32, 1024, lane, af, true);
    acc.v = wmma_bf16(af, bouts[c2], acc.v);
  }
#pragma unroll
  for (int r = 0; r < 8; r++) PD[wid*256 + (r + 8*hh)*16 + n] = acc.f[r];
  __syncthreads();
  if (wid == 0 && n < 8) {
#pragma unroll
    for (int r = 0; r < 8; r++) {
      int mr = r + 8*hh; int v = vbase + mr;
      float s = PD[mr*16+n] + PD[256+mr*16+n] + PD[512+mr*16+n] + PD[768+mr*16+n] + bm_out[n];
      dout[OFF_PM + (size_t)b*256 + n*32 + v] = s * weight[v] + meanall[v] * weight[32 + v];
    }
  }
}

// ---------------- fused sigma-branch MLP (WMMA, P = X^T X restructure) ----------------
__global__ __launch_bounds__(128) void k_sigma_branch(const float* bx, const float* bs_in_,
    const float* w1, const float* b1_, const float* w2, const float* b2_, const float* bs_out_,
    const float* wsf, const unsigned* wbU, float* dout) {
  extern __shared__ char smem[];
  float*    Xv  = (float*)smem;           // 28x12
  float*    Xq  = Xv + 336;               // 28x16x12
  unsigned* Pu  = (unsigned*)(Xq + 5376); // 16x80 uints (16x160 bf16)
  unsigned* Axs = Pu + 1280;              // 16x16 uints
  float*    Avt = (float*)(Axs + 256);    // 512
  float*    Cqt = Avt + 512;              // 16x512
  float*    h2  = Cqt + 8192;             // 16x1024
  float*    PD  = h2 + 16384;             // 4x16x16
  int blk = blockIdx.x; int b = blk / 64; int t64 = blk % 64;
  int v = t64 >> 1; int qbase = (t64 & 1) * 16;
  int tid = threadIdx.x, lane = tid & 31, wid = tid >> 5;
  for (int i = tid; i < 336; i += 128) {
    int m = i / 12, k = i % 12;
    Xv[i] = bx[(size_t)b*10752 + (m*12+k)*32 + v];
  }
  for (int i = tid; i < 5376; i += 128) {
    int m = i / 192, rem = i % 192, qi = rem / 12, k = rem % 12;
    Xq[(m*16 + qi)*12 + k] = bx[(size_t)b*10752 + (m*12+k)*32 + qbase + qi];
  }
  __syncthreads();
  for (int i = tid; i < 1280; i += 128) {          // P[qi][k1*12+k2] = sum_m Xv[m][k1] Xq[m][qi][k2]
    int qi = i / 80, kp = i % 80;
    float r0 = 0.f, r1 = 0.f;
    int kk0 = kp*2, kk1 = kk0 + 1;
    if (kk0 < 144) { int k1 = kk0/12, k2 = kk0%12; float s = 0.f;
      for (int m = 0; m < 28; m++) s += Xv[m*12+k1] * Xq[(m*16+qi)*12+k2]; r0 = s; }
    if (kk1 < 144) { int k1 = kk1/12, k2 = kk1%12; float s = 0.f;
      for (int m = 0; m < 28; m++) s += Xv[m*12+k1] * Xq[(m*16+qi)*12+k2]; r1 = s; }
    Pu[qi*80 + kp] = pack_bf16(r0, r1);
  }
  for (int i = tid; i < 256; i += 128) {           // xs rows: Sigma[b,m,q,v] over m (K pad 28->32)
    int qi = i >> 4, kp = i & 15;
    int k0 = kp*2, k1 = k0 + 1;
    float a = (k0 < 28) ? wsf[WSF_SIGMA + (size_t)(b*28+k0)*1024 + (qbase+qi)*32 + v] : 0.f;
    float c = (k1 < 28) ? wsf[WSF_SIGMA + (size_t)(b*28+k1)*1024 + (qbase+qi)*32 + v] : 0.f;
    Axs[qi*16 + kp] = pack_bf16(a, c);
  }
  const float* SxB = wsf + WSF_SX + b*384;
  for (int t = tid; t < 512; t += 128) {           // A_v(t) = Sx[v,:] . w1[t,:]
    float s = 0.f;
    for (int k = 0; k < 12; k++) s += SxB[v*12+k] * w1[t*12+k];
    Avt[t] = s;
  }
  for (int i = tid; i < 8192; i += 128) {          // C_q(t) = Sx[q,:] . w2[t,:]
    int qi = i >> 9, t = i & 511;
    float s = 0.f;
    for (int k = 0; k < 12; k++) s += SxB[(qbase+qi)*12+k] * w2[t*12+k];
    Cqt[qi*512 + t] = s;
  }
  __syncthreads();
  int n = lane & 15, hh = lane >> 4;
  // hoist nt-invariant A fragments
  unsigned afxs[8];
  load_fragA(Axs, 16, lane, afxs);
  unsigned afP[5][8];
#pragma unroll
  for (int kt = 0; kt < 5; kt++) load_fragA(Pu + kt*16, 80, lane, afP[kt]);
  for (int nt = wid; nt < 32; nt += 4) {
    // preload B fragments for this nt
    unsigned bfs[8];
    load_fragB(wbU + WBU_WSIN + nt*16*16, 16, lane, bfs);
    unsigned b12[5][8];
#pragma unroll
    for (int kt = 0; kt < 5; kt++)
      load_fragB(wbU + WBU_W12 + nt*16*80 + kt*16, 80, lane, b12[kt]);
    v8f_u acc; acc.v = (v8f){};
    acc.v = wmma_bf16(afxs, bfs, acc.v);
#pragma unroll
    for (int r = 0; r < 8; r++) {
      int qi = r + 8*hh, t = nt*16 + n;
      h2[qi*1024 + t] = tanhf(acc.f[r] + bs_in_[t]);
    }
    v8f_u acc2; acc2.v = (v8f){};
#pragma unroll
    for (int kt = 0; kt < 5; kt++)
      acc2.v = wmma_bf16(afP[kt], b12[kt], acc2.v);
#pragma unroll
    for (int r = 0; r < 8; r++) {
      int qi = r + 8*hh, t = nt*16 + n;
      float val = acc2.f[r] + b2_[t]*Avt[t] + b1_[t]*Cqt[qi*512+t] + 28.f*b1_[t]*b2_[t];
      h2[qi*1024 + 512 + t] = tanhf(val);
    }
  }
  __syncthreads();
  // output GEMM: h2 @ ws_out^T, split over 4 waves; B preloaded
  unsigned bouts[8][8];
#pragma unroll
  for (int c2 = 0; c2 < 8; c2++) {
    int kt = wid + 4*c2;
    load_fragB(wbU + WBU_WSOUT + kt*16, 512, lane, bouts[c2]);
  }
  v8f_u acc; acc.v = (v8f){};
#pragma unroll
  for (int c2 = 0; c2 < 8; c2++) {
    int kt = wid + 4*c2;
    unsigned af[8];
    load_fragA_f32(h2 + kt*32, 1024, lane, af, false);
    acc.v = wmma_bf16(af, bouts[c2], acc.v);
  }
#pragma unroll
  for (int r = 0; r < 8; r++) PD[wid*256 + (r + 8*hh)*16 + n] = acc.f[r];
  __syncthreads();
  if (wid == 0 && n < 8) {
#pragma unroll
    for (int r = 0; r < 8; r++) {
      int qi = r + 8*hh; int q = qbase + qi;
      float s = PD[qi*16+n] + PD[256+qi*16+n] + PD[512+qi*16+n] + PD[768+qi*16+n] + bs_out_[n];
      dout[OFF_PS + (size_t)b*8192 + n*1024 + v*32 + q] = fmaxf(s, 0.f);
    }
  }
}

// ---------------- launcher ----------------
extern "C" void kernel_launch(void* const* d_in, const int* in_sizes, int n_in,
                              void* d_out, int out_size, void* d_ws, size_t ws_size,
                              hipStream_t stream) {
  (void)in_sizes; (void)n_in; (void)out_size; (void)ws_size;
  const float* bx     = (const float*)d_in[0];
  const float* weight = (const float*)d_in[1];
  const float* wm_in  = (const float*)d_in[2];
  const float* bm_in  = (const float*)d_in[3];
  const float* wm_raw = (const float*)d_in[4];
  const float* bm_raw = (const float*)d_in[5];
  const float* wm_out = (const float*)d_in[6];
  const float* bm_out = (const float*)d_in[7];
  const float* ws_in  = (const float*)d_in[8];
  const float* bs_in  = (const float*)d_in[9];
  const float* ws_r1  = (const float*)d_in[10];
  const float* bs_r1  = (const float*)d_in[11];
  const float* ws_r2  = (const float*)d_in[12];
  const float* bs_r2  = (const float*)d_in[13];
  const float* ws_out = (const float*)d_in[14];
  const float* bs_out = (const float*)d_in[15];
  float* wsf = (float*)d_ws;
  unsigned short* wb = (unsigned short*)(wsf + WSF_END);
  const unsigned* wbU = (const unsigned*)wb;
  float* dout = (float*)d_out;

  k_prep_weights<<<64, 256, 0, stream>>>(wm_in, wm_raw, wm_out, ws_in, ws_r1, ws_r2, ws_out, wb);
  k_stats<<<7168, 128, 0, stream>>>(bx, wsf);
  k_meanall_sx<<<256, 128, 0, stream>>>(bx, wsf);
  k_eigen<<<7168, 32, 0, stream>>>(bx, wsf, dout);
  k_mean_branch<<<512, 128, 81920, stream>>>(bx, bm_in, bm_raw, bm_out, weight, wsf, wbU, dout);
  k_sigma_branch<<<16384, 128, 133440, stream>>>(bx, bs_in, ws_r1, bs_r1, ws_r2, bs_r2, bs_out,
                                                 wsf, wbU, dout);
}
